// GACNN_1022202216786
// MI455X (gfx1250) — matrix operand order repaired and checked by reading, not compiled, not run on hardware
//
#include <hip/hip_runtime.h>
#include <hip/hip_bf16.h>

typedef __attribute__((ext_vector_type(16))) __bf16 v16bf;
typedef __attribute__((ext_vector_type(8)))  __bf16 v8bf;
typedef __attribute__((ext_vector_type(8)))  float  v8f;
typedef __attribute__((ext_vector_type(4)))  float  f4;

#define N_NODES 100000
#define N_EDGES 1600000
#define F_IN    64
#define HEADS   4
#define F_OUT   16
#define HF      64          // HEADS * F_OUT
#define NEG_SLOPE 0.2f
#define BSTRIDE 72          // padded k-stride (144 B) to avoid LDS bank-period conflicts

// ---------------------------------------------------------------------------
// Kernel 0: init  out[n][c] = bias[c];  m = -FLT_MAX; denom = 0
// ---------------------------------------------------------------------------
__global__ void __launch_bounds__(256) gat_init(float* __restrict__ out,
                                                const float* __restrict__ bias,
                                                float* __restrict__ m,
                                                float* __restrict__ denom) {
  int idx = blockIdx.x * 256 + threadIdx.x;
  if (idx < N_NODES * HF) out[idx] = bias[idx & (HF - 1)];
  if (idx < N_NODES * HEADS) { m[idx] = -3.0e38f; denom[idx] = 0.0f; }
}

// ---------------------------------------------------------------------------
// Kernel 1: WMMA projection  h[N,64] = x[N,64] @ Wc[64,64]
//   Wc[k][n] = W[n>>4][k][n&15]   (W is [H][F_in][F_out])
//   LDS holds Wc TRANSPOSED: BsT[n][k] (k-contiguous, stride 72) so each lane's
//   16-K B fragment is two contiguous 16B reads -> ds_load_b128 x2.
//   one 16-row tile per wave; 8x v_wmma_f32_16x16x32_bf16 per wave.
// ---------------------------------------------------------------------------
__global__ void __launch_bounds__(256) gat_proj_wmma(const float* __restrict__ x,
                                                     const float* __restrict__ W,
                                                     float* __restrict__ h) {
  __shared__ __attribute__((aligned(16))) __bf16 BsT[HF * BSTRIDE];  // 9 KB
  for (int i = threadIdx.x; i < F_IN * HF; i += 256) {
    int k = i & 63, n = i >> 6;
    BsT[n * BSTRIDE + k] =
        (__bf16)W[(n >> 4) * (F_IN * F_OUT) + k * F_OUT + (n & 15)];
  }
  __syncthreads();

  const int wave = threadIdx.x >> 5;
  const int lane = threadIdx.x & 31;
  const int tile = blockIdx.x * 8 + wave;
  if (tile * 16 >= N_NODES) return;             // wave-uniform guard (EXEC stays full)

  const int rowBase = tile * 16;
  const int halfSel = lane >> 4;                // 0: lanes 0-15, 1: lanes 16-31
  const int l15     = lane & 15;
  const float* xr = x + (size_t)(rowBase + l15) * F_IN;

  // ---- B fragments ----------------------------------------------------------
  // lane<16: K = kt*32 + j (j=0..15); lane>=16: K = kt*32 + 16 + j; col = nt*16+l15
  // BsT row = col, contiguous K -> two aligned v8bf LDS loads per fragment.
  v16bf bfr[2][4];
#pragma unroll
  for (int kt = 0; kt < 2; ++kt) {
#pragma unroll
    for (int nt = 0; nt < 4; ++nt) {
      const __bf16* p = BsT + (nt * 16 + l15) * BSTRIDE + kt * 32 + halfSel * 16;
      v8bf lo = *(const v8bf*)(p);
      v8bf hi = *(const v8bf*)(p + 8);
      bfr[kt][nt] = __builtin_shufflevector(lo, hi, 0, 1, 2, 3, 4, 5, 6, 7,
                                            8, 9, 10, 11, 12, 13, 14, 15);
    }
  }

  v8f acc[4] = {};                              // zero accumulators
#pragma unroll
  for (int kt = 0; kt < 2; ++kt) {
    // A frag: j=0..15 -> K = kt*32 + halfSel*16 + j
    const int kb = kt * 32 + halfSel * 16;
    f4 a0 = *(const f4*)(xr + kb);
    f4 a1 = *(const f4*)(xr + kb + 4);
    f4 a2 = *(const f4*)(xr + kb + 8);
    f4 a3 = *(const f4*)(xr + kb + 12);
    v16bf a;
    a[0]=(__bf16)a0.x; a[1]=(__bf16)a0.y; a[2]=(__bf16)a0.z; a[3]=(__bf16)a0.w;
    a[4]=(__bf16)a1.x; a[5]=(__bf16)a1.y; a[6]=(__bf16)a1.z; a[7]=(__bf16)a1.w;
    a[8]=(__bf16)a2.x; a[9]=(__bf16)a2.y; a[10]=(__bf16)a2.z; a[11]=(__bf16)a2.w;
    a[12]=(__bf16)a3.x; a[13]=(__bf16)a3.y; a[14]=(__bf16)a3.z; a[15]=(__bf16)a3.w;
#pragma unroll
    for (int nt = 0; nt < 4; ++nt)
      acc[nt] = __builtin_amdgcn_wmma_f32_16x16x32_bf16(
          false, a, false, bfr[kt][nt], (short)0, acc[nt], false, false);
  }

  // C/D layout: row = v + halfSel*8, col = l15
#pragma unroll
  for (int nt = 0; nt < 4; ++nt) {
#pragma unroll
    for (int v = 0; v < 8; ++v)
      h[(size_t)(rowBase + v + halfSel * 8) * HF + nt * 16 + l15] = acc[nt][v];
  }
}

// ---------------------------------------------------------------------------
// Kernel 2: per-node logits  ls[n][hh] = h[n,hh,:].a_src[hh]; same for ld
// ---------------------------------------------------------------------------
__global__ void __launch_bounds__(256) gat_logits(const float* __restrict__ h,
                                                  const float* __restrict__ a_src,
                                                  const float* __restrict__ a_dst,
                                                  float* __restrict__ ls,
                                                  float* __restrict__ ld_) {
  int idx = blockIdx.x * 256 + threadIdx.x;
  if (idx >= N_NODES * HEADS) return;
  const int n = idx >> 2, hh = idx & 3;
  const float* hp = h + (size_t)n * HF + hh * F_OUT;
  const float* as = a_src + hh * F_OUT;
  const float* ad = a_dst + hh * F_OUT;
  float s = 0.f, d = 0.f;
#pragma unroll
  for (int o = 0; o < F_OUT; ++o) { float v = hp[o]; s += v * as[o]; d += v * ad[o]; }
  ls[idx] = s;
  ld_[idx] = d;
}

__device__ __forceinline__ float lrelu(float v) {
  return v > 0.f ? v : NEG_SLOPE * v;
}

// ---------------------------------------------------------------------------
// Kernel 3: segment max  m[dst][hh] = max over edges of lrelu(ls[src]+ld[dst])
// ---------------------------------------------------------------------------
__global__ void __launch_bounds__(256) gat_edge_max(const long long* __restrict__ ei,
                                                    const float* __restrict__ ls,
                                                    const float* __restrict__ ld_,
                                                    float* __restrict__ m) {
  int e = blockIdx.x * 256 + threadIdx.x;
  if (e >= N_EDGES) return;
  const long long s = ei[e], d = ei[N_EDGES + e];
  const f4 lsv = *(const f4*)(ls + s * HEADS);
  const f4 ldv = *(const f4*)(ld_ + d * HEADS);
#pragma unroll
  for (int hh = 0; hh < HEADS; ++hh)
    atomicMax(m + d * HEADS + hh, lrelu(lsv[hh] + ldv[hh]));
}

// ---------------------------------------------------------------------------
// Kernel 4: denom[dst][hh] += exp(e - m[dst][hh])
// ---------------------------------------------------------------------------
__global__ void __launch_bounds__(256) gat_edge_expsum(const long long* __restrict__ ei,
                                                       const float* __restrict__ ls,
                                                       const float* __restrict__ ld_,
                                                       const float* __restrict__ m,
                                                       float* __restrict__ denom) {
  int e = blockIdx.x * 256 + threadIdx.x;
  if (e >= N_EDGES) return;
  const long long s = ei[e], d = ei[N_EDGES + e];
  const f4 lsv = *(const f4*)(ls + s * HEADS);
  const f4 ldv = *(const f4*)(ld_ + d * HEADS);
  const f4 mv  = *(const f4*)(m + d * HEADS);
#pragma unroll
  for (int hh = 0; hh < HEADS; ++hh)
    atomicAdd(denom + d * HEADS + hh, __expf(lrelu(lsv[hh] + ldv[hh]) - mv[hh]));
}

// ---------------------------------------------------------------------------
// Kernel 5: out[dst] += alpha * h[src]   (64 float atomics / edge, L2-resident)
// ---------------------------------------------------------------------------
__global__ void __launch_bounds__(256) gat_edge_aggregate(const long long* __restrict__ ei,
                                                          const float* __restrict__ ls,
                                                          const float* __restrict__ ld_,
                                                          const float* __restrict__ m,
                                                          const float* __restrict__ denom,
                                                          const float* __restrict__ h,
                                                          float* __restrict__ out) {
  int e = blockIdx.x * 256 + threadIdx.x;
  if (e >= N_EDGES) return;
  const long long s = ei[e], d = ei[N_EDGES + e];
  const f4 lsv = *(const f4*)(ls + s * HEADS);
  const f4 ldv = *(const f4*)(ld_ + d * HEADS);
  const f4 mv  = *(const f4*)(m + d * HEADS);
  const f4 dnv = *(const f4*)(denom + d * HEADS);
  const f4* hs = (const f4*)(h + s * HF);
  float* od = out + d * HF;
#pragma unroll
  for (int hh = 0; hh < HEADS; ++hh) {
    const float alpha = __expf(lrelu(lsv[hh] + ldv[hh]) - mv[hh]) / (dnv[hh] + 1e-16f);
#pragma unroll
    for (int q = 0; q < 4; ++q) {
      const f4 hv = hs[hh * 4 + q];
      atomicAdd(od + hh * F_OUT + q * 4 + 0, alpha * hv.x);
      atomicAdd(od + hh * F_OUT + q * 4 + 1, alpha * hv.y);
      atomicAdd(od + hh * F_OUT + q * 4 + 2, alpha * hv.z);
      atomicAdd(od + hh * F_OUT + q * 4 + 3, alpha * hv.w);
    }
  }
}

// ---------------------------------------------------------------------------
extern "C" void kernel_launch(void* const* d_in, const int* in_sizes, int n_in,
                              void* d_out, int out_size, void* d_ws, size_t ws_size,
                              hipStream_t stream) {
  const float*     x     = (const float*)d_in[0];
  const long long* eidx  = (const long long*)d_in[1];   // int64 [2, E]
  const float*     W     = (const float*)d_in[2];       // [H, F_in, F_out]
  const float*     a_src = (const float*)d_in[3];
  const float*     a_dst = (const float*)d_in[4];
  const float*     bias  = (const float*)d_in[5];
  float* out = (float*)d_out;

  // workspace layout (floats): h | ls | ld | m | denom  = 32 MB total
  float* ws    = (float*)d_ws;
  float* h     = ws;                                    // N*64
  float* ls    = h  + (size_t)N_NODES * HF;             // N*4
  float* ld_   = ls + (size_t)N_NODES * HEADS;          // N*4
  float* m     = ld_ + (size_t)N_NODES * HEADS;         // N*4
  float* denom = m  + (size_t)N_NODES * HEADS;          // N*4

  const int initBlocks  = (N_NODES * HF + 255) / 256;
  const int projBlocks  = (N_NODES / 16 + 7) / 8;       // 1 wave per 16-row tile
  const int logitBlocks = (N_NODES * HEADS + 255) / 256;
  const int edgeBlocks  = (N_EDGES + 255) / 256;

  gat_init<<<initBlocks, 256, 0, stream>>>(out, bias, m, denom);
  gat_proj_wmma<<<projBlocks, 256, 0, stream>>>(x, W, h);
  gat_logits<<<logitBlocks, 256, 0, stream>>>(h, a_src, a_dst, ls, ld_);
  gat_edge_max<<<edgeBlocks, 256, 0, stream>>>(eidx, ls, ld_, m);
  gat_edge_expsum<<<edgeBlocks, 256, 0, stream>>>(eidx, ls, ld_, m, denom);
  gat_edge_aggregate<<<edgeBlocks, 256, 0, stream>>>(eidx, ls, ld_, m, denom, h, out);
}
